// EncoderDecoderLSTM_65300682768674
// MI455X (gfx1250) — matrix operand order repaired and checked by reading, not compiled, not run on hardware
//
#include <hip/hip_runtime.h>

// Problem constants (match reference)
#define BATCH   4096
#define TSEQ    365
#define HID     64
#define GATES   256      // 4*H
#define HOR     7
#define WAVES   4
#define TPB     (WAVES * 32)
#define TILES   (BATCH / 16)      // 256 batch tiles of 16 rows
#define BLOCKS  (TILES / WAVES)   // 64 blocks

typedef __attribute__((ext_vector_type(16))) _Float16 v16h;
typedef __attribute__((ext_vector_type(8)))  float    v8f;

__device__ __forceinline__ v8f wmma16(v16h a, v16h b, v8f c) {
  // D = A(16x32 f16) * B(32x16 f16) + C(16x16 f32)
  return __builtin_amdgcn_wmma_f32_16x16x32_f16(false, a, false, b, (short)0, c,
                                                false, false);
}

// Branchless activations on the CDNA5 transcendental pipe (v_tanh_f32).
__device__ __forceinline__ float fast_tanh(float x) {
#if __has_builtin(__builtin_amdgcn_tanhf)
  return __builtin_amdgcn_tanhf(x);
#else
  float e = __expf(x + x);                       // v_exp_f32 (native)
  return 1.f - 2.f * __builtin_amdgcn_rcpf(e + 1.f);
#endif
}
__device__ __forceinline__ float sigm(float x) {
  return 0.5f + 0.5f * fast_tanh(0.5f * x);      // branchless sigmoid
}

// Address (in halves) of element (m, kcol) inside a 16x64 f16 A-fragment
// buffer laid out so that lane L reads its v16h at [kc*512 + L*16].
// A-layout (16-bit, 16x32): lanes 0-15 hold K 0-7 / 16-23, lanes 16-31 hold
// K 8-15 / 24-31, two K values per VGPR.
__device__ __forceinline__ int afrag_addr(int m, int kcol) {
  int kc = kcol >> 5;
  int kl = kcol & 31;
  int hi, j;
  if (kl < 8)       { hi = 0; j = kl;       }
  else if (kl < 16) { hi = 1; j = kl - 8;   }
  else if (kl < 24) { hi = 0; j = kl - 8;   }
  else              { hi = 1; j = kl - 16;  }
  return kc * 512 + (hi * 16 + m) * 16 + j;
}

// Stage a 256x64 row-major f32 weight matrix (gates x K) into LDS as 32
// WMMA B-fragments (16 N-tiles x 2 K-chunks), f16, so that lane L of a wave
// reads its v16h operand contiguously at [(nt*2+kc)*512 + L*16].
// B-layout (16-bit, 32x16): lanes 0-15 hold K 0-7 / 16-23 of the chunk, lanes
// 16-31 hold K 8-15 / 24-31; n = L%16.
__device__ __forceinline__ void stage_bmat(const float* __restrict__ W,
                                           _Float16* dst, int tid) {
  for (int idx = tid; idx < 16384; idx += TPB) {
    int f  = idx >> 9;        // fragment 0..31
    int r  = idx & 511;
    int L  = r >> 4;
    int j  = r & 15;
    int nt = f >> 1, kc = f & 1;
    int n  = L & 15, hi = L >> 4;
    int k  = kc * 32 + hi * 8 + j + ((j >> 3) << 3);
    int g  = nt * 16 + n;
    dst[idx] = (_Float16)W[g * HID + k];
  }
}

// ---------------------------------------------------------------------------
// Encoder: 2-layer LSTM over TSEQ steps; writes final (h0,h1 as A-frags,
// c0,c1 as f32 C-layout) per batch tile into the workspace.
// Workspace layout per tile (12288 bytes):
//   [0,2048)    h0 A-frags (f16)   [2048,4096)  h1 A-frags (f16)
//   [4096,8192) c0 (f32)           [8192,12288) c1 (f32)
// ---------------------------------------------------------------------------
__global__ __launch_bounds__(TPB) void enc_lstm_kernel(
    const float* __restrict__ x,
    const float* __restrict__ Wih0, const float* __restrict__ Whh0,
    const float* __restrict__ b0,
    const float* __restrict__ Wih1, const float* __restrict__ Whh1,
    const float* __restrict__ b1,
    char* __restrict__ ws) {
  __shared__ alignas(32) _Float16 sW[3 * 16384];           // 96 KB
  __shared__ float sB0[GATES], sB1[GATES], sI0[GATES];
  __shared__ alignas(32) _Float16 sH[WAVES][2][1024];      // per-wave h scratch

  int tid = threadIdx.x;
  stage_bmat(Whh0, sW,         tid);
  stage_bmat(Wih1, sW + 16384, tid);
  stage_bmat(Whh1, sW + 32768, tid);
  for (int i = tid; i < GATES; i += TPB) {
    sB0[i] = b0[i]; sB1[i] = b1[i]; sI0[i] = Wih0[i];
  }
  __syncthreads();

  int lane = tid & 31;
  int wave = tid >> 5;
  int n  = lane & 15, hi = lane >> 4;
  int tile = blockIdx.x * WAVES + wave;
  int brow = tile * 16;

  _Float16* h0s = &sH[wave][0][0];
  _Float16* h1s = &sH[wave][1][0];
  const _Float16* W0  = sW;
  const _Float16* W1i = sW + 16384;
  const _Float16* W1h = sW + 32768;

  v16h a0[2] = {}, a1[2] = {};
  v8f  c0[4] = {}, c1[4] = {};

  // Hoist per-lane bias / input-weight scalars (gate order i,f,g,o).
  float bi0[4], bf0[4], bg0[4], bo0[4];
  float wi0[4], wf0[4], wg0[4], wo0[4];
  float bi1[4], bf1[4], bg1[4], bo1[4];
#pragma unroll
  for (int cg = 0; cg < 4; ++cg) {
    bi0[cg] = sB0[(cg + 0)  * 16 + n];  wi0[cg] = sI0[(cg + 0)  * 16 + n];
    bf0[cg] = sB0[(cg + 4)  * 16 + n];  wf0[cg] = sI0[(cg + 4)  * 16 + n];
    bg0[cg] = sB0[(cg + 8)  * 16 + n];  wg0[cg] = sI0[(cg + 8)  * 16 + n];
    bo0[cg] = sB0[(cg + 12) * 16 + n];  wo0[cg] = sI0[(cg + 12) * 16 + n];
    bi1[cg] = sB1[(cg + 0)  * 16 + n];
    bf1[cg] = sB1[(cg + 4)  * 16 + n];
    bg1[cg] = sB1[(cg + 8)  * 16 + n];
    bo1[cg] = sB1[(cg + 12) * 16 + n];
  }

  for (int t = 0; t < TSEQ; ++t) {
    float xv[8];
#pragma unroll
    for (int r = 0; r < 8; ++r)
      xv[r] = x[(brow + r + 8 * hi) * TSEQ + t];

    // ---------------- layer 0 ----------------
#pragma unroll
    for (int cg = 0; cg < 4; ++cg) {
      v8f gi, gf, gg, go;
#pragma unroll
      for (int r = 0; r < 8; ++r) {
        gi[r] = bi0[cg] + xv[r] * wi0[cg];
        gf[r] = bf0[cg] + xv[r] * wf0[cg];
        gg[r] = bg0[cg] + xv[r] * wg0[cg];
        go[r] = bo0[cg] + xv[r] * wo0[cg];
      }
#pragma unroll
      for (int kc = 0; kc < 2; ++kc) {
        gi = wmma16(a0[kc], *(const v16h*)&W0[((cg + 0)  * 2 + kc) * 512 + lane * 16], gi);
        gf = wmma16(a0[kc], *(const v16h*)&W0[((cg + 4)  * 2 + kc) * 512 + lane * 16], gf);
        gg = wmma16(a0[kc], *(const v16h*)&W0[((cg + 8)  * 2 + kc) * 512 + lane * 16], gg);
        go = wmma16(a0[kc], *(const v16h*)&W0[((cg + 12) * 2 + kc) * 512 + lane * 16], go);
      }
#pragma unroll
      for (int r = 0; r < 8; ++r) {
        float cn = sigm(gf[r]) * c0[cg][r] + sigm(gi[r]) * fast_tanh(gg[r]);
        c0[cg][r] = cn;
        float h = sigm(go[r]) * fast_tanh(cn);
        h0s[afrag_addr(r + 8 * hi, cg * 16 + n)] = (_Float16)h;   // scatter in A-frag order
      }
    }
    a0[0] = *(const v16h*)&h0s[lane * 16];          // same-wave LDS: in order
    a0[1] = *(const v16h*)&h0s[512 + lane * 16];

    // ---------------- layer 1 ----------------
#pragma unroll
    for (int cg = 0; cg < 4; ++cg) {
      v8f gi, gf, gg, go;
#pragma unroll
      for (int r = 0; r < 8; ++r) {
        gi[r] = bi1[cg]; gf[r] = bf1[cg]; gg[r] = bg1[cg]; go[r] = bo1[cg];
      }
#pragma unroll
      for (int kc = 0; kc < 2; ++kc) {
        gi = wmma16(a0[kc], *(const v16h*)&W1i[((cg + 0)  * 2 + kc) * 512 + lane * 16], gi);
        gf = wmma16(a0[kc], *(const v16h*)&W1i[((cg + 4)  * 2 + kc) * 512 + lane * 16], gf);
        gg = wmma16(a0[kc], *(const v16h*)&W1i[((cg + 8)  * 2 + kc) * 512 + lane * 16], gg);
        go = wmma16(a0[kc], *(const v16h*)&W1i[((cg + 12) * 2 + kc) * 512 + lane * 16], go);
        gi = wmma16(a1[kc], *(const v16h*)&W1h[((cg + 0)  * 2 + kc) * 512 + lane * 16], gi);
        gf = wmma16(a1[kc], *(const v16h*)&W1h[((cg + 4)  * 2 + kc) * 512 + lane * 16], gf);
        gg = wmma16(a1[kc], *(const v16h*)&W1h[((cg + 8)  * 2 + kc) * 512 + lane * 16], gg);
        go = wmma16(a1[kc], *(const v16h*)&W1h[((cg + 12) * 2 + kc) * 512 + lane * 16], go);
      }
#pragma unroll
      for (int r = 0; r < 8; ++r) {
        float cn = sigm(gf[r]) * c1[cg][r] + sigm(gi[r]) * fast_tanh(gg[r]);
        c1[cg][r] = cn;
        float h = sigm(go[r]) * fast_tanh(cn);
        h1s[afrag_addr(r + 8 * hi, cg * 16 + n)] = (_Float16)h;
      }
    }
    a1[0] = *(const v16h*)&h1s[lane * 16];
    a1[1] = *(const v16h*)&h1s[512 + lane * 16];
  }

  // Dump final state (layout keyed only by (tile, lane) -> decoder reloads).
  char* base = ws + (size_t)tile * 12288;
  _Float16* h0w = (_Float16*)base;
  _Float16* h1w = (_Float16*)(base + 2048);
  float*    c0w = (float*)(base + 4096);
  float*    c1w = (float*)(base + 8192);
#pragma unroll
  for (int kc = 0; kc < 2; ++kc) {
    *(v16h*)&h0w[kc * 512 + lane * 16] = a0[kc];
    *(v16h*)&h1w[kc * 512 + lane * 16] = a1[kc];
  }
#pragma unroll
  for (int cg = 0; cg < 4; ++cg) {
    *(v8f*)&c0w[cg * 256 + lane * 8] = c0[cg];
    *(v8f*)&c1w[cg * 256 + lane * 8] = c1[cg];
  }
}

// ---------------------------------------------------------------------------
// Decoder: 7 autoregressive steps; linear head; writes [B, 7] f32.
// ---------------------------------------------------------------------------
__global__ __launch_bounds__(TPB) void dec_lstm_kernel(
    const float* __restrict__ Wih0, const float* __restrict__ Whh0,
    const float* __restrict__ b0,
    const float* __restrict__ Wih1, const float* __restrict__ Whh1,
    const float* __restrict__ b1,
    const float* __restrict__ fcW, const float* __restrict__ fcB,
    const char* __restrict__ ws, float* __restrict__ out) {
  __shared__ alignas(32) _Float16 sW[3 * 16384];
  __shared__ float sB0[GATES], sB1[GATES], sI0[GATES], sFc[HID];
  __shared__ alignas(32) _Float16 sH[WAVES][2][1024];

  int tid = threadIdx.x;
  stage_bmat(Whh0, sW,         tid);
  stage_bmat(Wih1, sW + 16384, tid);
  stage_bmat(Whh1, sW + 32768, tid);
  for (int i = tid; i < GATES; i += TPB) {
    sB0[i] = b0[i]; sB1[i] = b1[i]; sI0[i] = Wih0[i];
  }
  for (int i = tid; i < HID; i += TPB) sFc[i] = fcW[i];
  __syncthreads();

  int lane = tid & 31;
  int wave = tid >> 5;
  int n  = lane & 15, hi = lane >> 4;
  int tile = blockIdx.x * WAVES + wave;
  int brow = tile * 16;

  _Float16* h0s = &sH[wave][0][0];
  _Float16* h1s = &sH[wave][1][0];
  const _Float16* W0  = sW;
  const _Float16* W1i = sW + 16384;
  const _Float16* W1h = sW + 32768;

  // Restore encoder-final state.
  const char* base = ws + (size_t)tile * 12288;
  const _Float16* h0w = (const _Float16*)base;
  const _Float16* h1w = (const _Float16*)(base + 2048);
  const float*    c0w = (const float*)(base + 4096);
  const float*    c1w = (const float*)(base + 8192);
  v16h a0[2], a1[2];
  v8f  c0[4], c1[4];
#pragma unroll
  for (int kc = 0; kc < 2; ++kc) {
    a0[kc] = *(const v16h*)&h0w[kc * 512 + lane * 16];
    a1[kc] = *(const v16h*)&h1w[kc * 512 + lane * 16];
  }
#pragma unroll
  for (int cg = 0; cg < 4; ++cg) {
    c0[cg] = *(const v8f*)&c0w[cg * 256 + lane * 8];
    c1[cg] = *(const v8f*)&c1w[cg * 256 + lane * 8];
  }

  float bi0[4], bf0[4], bg0[4], bo0[4];
  float wi0[4], wf0[4], wg0[4], wo0[4];
  float bi1[4], bf1[4], bg1[4], bo1[4], fcw[4];
#pragma unroll
  for (int cg = 0; cg < 4; ++cg) {
    bi0[cg] = sB0[(cg + 0)  * 16 + n];  wi0[cg] = sI0[(cg + 0)  * 16 + n];
    bf0[cg] = sB0[(cg + 4)  * 16 + n];  wf0[cg] = sI0[(cg + 4)  * 16 + n];
    bg0[cg] = sB0[(cg + 8)  * 16 + n];  wg0[cg] = sI0[(cg + 8)  * 16 + n];
    bo0[cg] = sB0[(cg + 12) * 16 + n];  wo0[cg] = sI0[(cg + 12) * 16 + n];
    bi1[cg] = sB1[(cg + 0)  * 16 + n];
    bf1[cg] = sB1[(cg + 4)  * 16 + n];
    bg1[cg] = sB1[(cg + 8)  * 16 + n];
    bo1[cg] = sB1[(cg + 12) * 16 + n];
    fcw[cg] = sFc[cg * 16 + n];
  }
  float fcb = fcB[0];

  float din[8];
#pragma unroll
  for (int r = 0; r < 8; ++r) din[r] = 0.f;

  for (int st = 0; st < HOR; ++st) {
    // ---------------- layer 0 ----------------
#pragma unroll
    for (int cg = 0; cg < 4; ++cg) {
      v8f gi, gf, gg, go;
#pragma unroll
      for (int r = 0; r < 8; ++r) {
        gi[r] = bi0[cg] + din[r] * wi0[cg];
        gf[r] = bf0[cg] + din[r] * wf0[cg];
        gg[r] = bg0[cg] + din[r] * wg0[cg];
        go[r] = bo0[cg] + din[r] * wo0[cg];
      }
#pragma unroll
      for (int kc = 0; kc < 2; ++kc) {
        gi = wmma16(a0[kc], *(const v16h*)&W0[((cg + 0)  * 2 + kc) * 512 + lane * 16], gi);
        gf = wmma16(a0[kc], *(const v16h*)&W0[((cg + 4)  * 2 + kc) * 512 + lane * 16], gf);
        gg = wmma16(a0[kc], *(const v16h*)&W0[((cg + 8)  * 2 + kc) * 512 + lane * 16], gg);
        go = wmma16(a0[kc], *(const v16h*)&W0[((cg + 12) * 2 + kc) * 512 + lane * 16], go);
      }
#pragma unroll
      for (int r = 0; r < 8; ++r) {
        float cn = sigm(gf[r]) * c0[cg][r] + sigm(gi[r]) * fast_tanh(gg[r]);
        c0[cg][r] = cn;
        float h = sigm(go[r]) * fast_tanh(cn);
        h0s[afrag_addr(r + 8 * hi, cg * 16 + n)] = (_Float16)h;
      }
    }
    a0[0] = *(const v16h*)&h0s[lane * 16];
    a0[1] = *(const v16h*)&h0s[512 + lane * 16];

    // ---------------- layer 1 ----------------
    v8f hsave[4];
#pragma unroll
    for (int cg = 0; cg < 4; ++cg) {
      v8f gi, gf, gg, go;
#pragma unroll
      for (int r = 0; r < 8; ++r) {
        gi[r] = bi1[cg]; gf[r] = bf1[cg]; gg[r] = bg1[cg]; go[r] = bo1[cg];
      }
#pragma unroll
      for (int kc = 0; kc < 2; ++kc) {
        gi = wmma16(a0[kc], *(const v16h*)&W1i[((cg + 0)  * 2 + kc) * 512 + lane * 16], gi);
        gf = wmma16(a0[kc], *(const v16h*)&W1i[((cg + 4)  * 2 + kc) * 512 + lane * 16], gf);
        gg = wmma16(a0[kc], *(const v16h*)&W1i[((cg + 8)  * 2 + kc) * 512 + lane * 16], gg);
        go = wmma16(a0[kc], *(const v16h*)&W1i[((cg + 12) * 2 + kc) * 512 + lane * 16], go);
        gi = wmma16(a1[kc], *(const v16h*)&W1h[((cg + 0)  * 2 + kc) * 512 + lane * 16], gi);
        gf = wmma16(a1[kc], *(const v16h*)&W1h[((cg + 4)  * 2 + kc) * 512 + lane * 16], gf);
        gg = wmma16(a1[kc], *(const v16h*)&W1h[((cg + 8)  * 2 + kc) * 512 + lane * 16], gg);
        go = wmma16(a1[kc], *(const v16h*)&W1h[((cg + 12) * 2 + kc) * 512 + lane * 16], go);
      }
#pragma unroll
      for (int r = 0; r < 8; ++r) {
        float cn = sigm(gf[r]) * c1[cg][r] + sigm(gi[r]) * fast_tanh(gg[r]);
        c1[cg][r] = cn;
        float h = sigm(go[r]) * fast_tanh(cn);
        hsave[cg][r] = h;
        h1s[afrag_addr(r + 8 * hi, cg * 16 + n)] = (_Float16)h;
      }
    }
    a1[0] = *(const v16h*)&h1s[lane * 16];
    a1[1] = *(const v16h*)&h1s[512 + lane * 16];

    // ---------------- fc head + feedback ----------------
#pragma unroll
    for (int r = 0; r < 8; ++r) {
      float s = hsave[0][r] * fcw[0] + hsave[1][r] * fcw[1] +
                hsave[2][r] * fcw[2] + hsave[3][r] * fcw[3];
      s += __shfl_xor(s, 1, 16);
      s += __shfl_xor(s, 2, 16);
      s += __shfl_xor(s, 4, 16);
      s += __shfl_xor(s, 8, 16);
      float p = s + fcb;
      din[r] = p;
      if (n == 0) out[(brow + r + 8 * hi) * HOR + st] = p;
    }
  }
}

extern "C" void kernel_launch(void* const* d_in, const int* in_sizes, int n_in,
                              void* d_out, int out_size, void* d_ws, size_t ws_size,
                              hipStream_t stream) {
  (void)in_sizes; (void)n_in; (void)out_size; (void)ws_size;
  const float* x      = (const float*)d_in[0];
  const float* eWih0  = (const float*)d_in[1];
  const float* eWhh0  = (const float*)d_in[2];
  const float* eb0    = (const float*)d_in[3];
  const float* eWih1  = (const float*)d_in[4];
  const float* eWhh1  = (const float*)d_in[5];
  const float* eb1    = (const float*)d_in[6];
  const float* dWih0  = (const float*)d_in[7];
  const float* dWhh0  = (const float*)d_in[8];
  const float* db0    = (const float*)d_in[9];
  const float* dWih1  = (const float*)d_in[10];
  const float* dWhh1  = (const float*)d_in[11];
  const float* db1    = (const float*)d_in[12];
  const float* fcW    = (const float*)d_in[13];
  const float* fcB    = (const float*)d_in[14];

  enc_lstm_kernel<<<BLOCKS, TPB, 0, stream>>>(x, eWih0, eWhh0, eb0, eWih1,
                                              eWhh1, eb1, (char*)d_ws);
  dec_lstm_kernel<<<BLOCKS, TPB, 0, stream>>>(dWih0, dWhh0, db0, dWih1, dWhh1,
                                              db1, fcW, fcB,
                                              (const char*)d_ws, (float*)d_out);
}
